// Probabilistic_DAG_Generator_Sinkhorn_79594333929922
// MI455X (gfx1250) — matrix operand order repaired and checked by reading, not compiled, not run on hardware
//
#include <hip/hip_runtime.h>

typedef __attribute__((ext_vector_type(16))) _Float16 v16h;
typedef __attribute__((ext_vector_type(8)))  _Float16 v8h;
typedef __attribute__((ext_vector_type(8)))  float    v8f;

#define N 4096
#define N2 (4096l*4096l)

// ---------------- utility fills ----------------
__global__ void fill_zero_f32(float* __restrict__ p, int n) {
  int i = blockIdx.x * blockDim.x + threadIdx.x;
  for (; i < n; i += gridDim.x * blockDim.x) p[i] = 0.0f;
}
__global__ void fill_zero_u128(uint4* __restrict__ p, long n) {
  long i = (long)blockIdx.x * blockDim.x + threadIdx.x;
  long stride = (long)gridDim.x * blockDim.x;
  uint4 z = {0u, 0u, 0u, 0u};
  for (; i < n; i += stride) p[i] = z;
}

// ---------------- la0 = pw + gumbel ----------------
__global__ void init_la(const float* __restrict__ pw, const float* __restrict__ g,
                        float* __restrict__ la) {
  long i = (long)blockIdx.x * blockDim.x + threadIdx.x;
  long stride = (long)gridDim.x * blockDim.x;
  for (; i < N2; i += stride) la[i] = pw[i] + g[i];
}

// ---------------- factored Sinkhorn ----------------
__device__ __forceinline__ void lse_combine(float& m1, float& s1, float m2, float s2) {
  if (m1 == -INFINITY && m2 == -INFINITY) { s1 = 0.0f; return; }
  float M = fmaxf(m1, m2);
  s1 = s1 * __expf(m1 - M) + s2 * __expf(m2 - M);
  m1 = M;
}

__global__ __launch_bounds__(256) void row_lse(const float* __restrict__ la,
                                               const float* __restrict__ C,
                                               float* __restrict__ R) {
  int i = blockIdx.x;
  const float* row = la + (size_t)i * N;
  float m = -INFINITY, s = 0.0f;
  for (int j = threadIdx.x; j < N; j += 256) {
    float x = row[j] - C[j];
    if (x > m) { s = s * __expf(m - x) + 1.0f; m = x; }
    else       { s += __expf(x - m); }
  }
  __shared__ float sm[256], ss[256];
  sm[threadIdx.x] = m; ss[threadIdx.x] = s;
  __syncthreads();
  for (int off = 128; off; off >>= 1) {
    if ((int)threadIdx.x < off) {
      float m1 = sm[threadIdx.x], s1 = ss[threadIdx.x];
      lse_combine(m1, s1, sm[threadIdx.x + off], ss[threadIdx.x + off]);
      sm[threadIdx.x] = m1; ss[threadIdx.x] = s1;
    }
    __syncthreads();
  }
  if (threadIdx.x == 0) R[i] = sm[0] + __logf(ss[0]);
}

// C[j] = lse_i(la[i,j]-R[i]) : chunked over 32 row blocks for parallelism
__global__ __launch_bounds__(256) void col_lse_part(const float* __restrict__ la,
                                                    const float* __restrict__ R,
                                                    float* __restrict__ Mp,
                                                    float* __restrict__ Sp) {
  int j = blockIdx.x * 256 + threadIdx.x;   // gridDim.x = 16
  int chunk = blockIdx.y;                   // 32 chunks x 128 rows
  float m = -INFINITY, s = 0.0f;
  int i0 = chunk * 128;
  for (int i = i0; i < i0 + 128; ++i) {
    float x = la[(size_t)i * N + j] - R[i];
    if (x > m) { s = s * __expf(m - x) + 1.0f; m = x; }
    else       { s += __expf(x - m); }
  }
  Mp[(size_t)chunk * N + j] = m;
  Sp[(size_t)chunk * N + j] = s;
}
__global__ __launch_bounds__(256) void col_lse_combine(const float* __restrict__ Mp,
                                                       const float* __restrict__ Sp,
                                                       float* __restrict__ C) {
  int j = blockIdx.x * 256 + threadIdx.x;
  float m = -INFINITY, s = 0.0f;
  for (int c = 0; c < 32; ++c)
    lse_combine(m, s, Mp[(size_t)c * N + j], Sp[(size_t)c * N + j]);
  C[j] = m + __logf(s);
}

// ---------------- hard permutation: row argmax of (la - C) ----------------
__global__ __launch_bounds__(256) void row_argmax(const float* __restrict__ la,
                                                  const float* __restrict__ C,
                                                  int* __restrict__ idx) {
  int i = blockIdx.x;
  const float* row = la + (size_t)i * N;
  float best = -INFINITY; int bj = N;
  for (int j = threadIdx.x; j < N; j += 256) {
    float v = row[j] - C[j];
    if (v > best) { best = v; bj = j; }   // ascending j per thread -> first max kept
  }
  __shared__ float sv[256]; __shared__ int si[256];
  sv[threadIdx.x] = best; si[threadIdx.x] = bj;
  __syncthreads();
  for (int off = 128; off; off >>= 1) {
    if ((int)threadIdx.x < off) {
      float v2 = sv[threadIdx.x + off]; int j2 = si[threadIdx.x + off];
      if (v2 > sv[threadIdx.x] || (v2 == sv[threadIdx.x] && j2 < si[threadIdx.x])) {
        sv[threadIdx.x] = v2; si[threadIdx.x] = j2;
      }
    }
    __syncthreads();
  }
  if (threadIdx.x == 0) idx[i] = si[0];
}

__global__ void build_p(const int* __restrict__ idx, _Float16* __restrict__ P,
                        _Float16* __restrict__ PT) {
  int i = blockIdx.x * 256 + threadIdx.x;
  if (i < N) {
    int a = idx[i];
    P[(size_t)i * N + a]  = (_Float16)1.0f;
    PT[(size_t)a * N + i] = (_Float16)1.0f;
  }
}

// ---------------- WMMA GEMM: Cout = A(f16) @ B ----------------
// EPI: 0 = store f16, 1 = store f16 with strict-upper-triangle mask, 2 = store f32
// BF32: B matrix is f32 in memory (converted on load)
// Block = 128 threads (4 waves), tile 64(M) x 64(N), K-step 32.
// B tile staged coalesced global->LDS transposed ([n][k], padded), fragments read
// as two 16B LDS vector loads; next tile's global loads overlap current WMMAs.
template <int EPI, bool BF32>
__global__ __launch_bounds__(128) void gemm_wmma(const _Float16* __restrict__ A,
                                                 const void* __restrict__ Bv,
                                                 void* __restrict__ Cv) {
  __shared__ __align__(16) _Float16 Bs[64][40];   // [n][k] transposed, 80B rows (16B aligned)
  const int tid  = threadIdx.x;
  const int lane = tid & 31;
  const int wave = tid >> 5;        // 4 waves stacked along M
  const int hsel = lane >> 4;
  const int l15  = lane & 15;
  const int m0 = blockIdx.y * 64 + wave * 16;
  const int n0 = blockIdx.x * 64;
  const _Float16* arow = A + (size_t)(m0 + l15) * N;

  // staging: 32x64 halves = 256 8-half chunks; 2 chunks per thread
  const int kr0 = tid >> 3,           cc0 = (tid & 7) * 8;
  const int kr1 = (tid + 128) >> 3,   cc1 = (tid & 7) * 8;

  v8f acc[4] = {};
  v8h g0, g1;

  auto load_chunks = [&](int k0) {
    if (BF32) {
      const float* B = (const float*)Bv;
      const float* p0 = B + (size_t)(k0 + kr0) * N + n0 + cc0;
      const float* p1 = B + (size_t)(k0 + kr1) * N + n0 + cc1;
#pragma unroll
      for (int i = 0; i < 8; ++i) { g0[i] = (_Float16)p0[i]; g1[i] = (_Float16)p1[i]; }
    } else {
      const _Float16* B = (const _Float16*)Bv;
      g0 = *reinterpret_cast<const v8h*>(B + (size_t)(k0 + kr0) * N + n0 + cc0);
      g1 = *reinterpret_cast<const v8h*>(B + (size_t)(k0 + kr1) * N + n0 + cc1);
    }
  };

  load_chunks(0);
  for (int k0 = 0; k0 < N; k0 += 32) {
    // commit staged chunks into LDS (transposed)
#pragma unroll
    for (int i = 0; i < 8; ++i) { Bs[cc0 + i][kr0] = g0[i]; Bs[cc1 + i][kr1] = g1[i]; }
    __syncthreads();
    if (k0 + 32 < N) load_chunks(k0 + 32);   // overlap next tile's global loads

    // ---- A fragment: lane=M, e<8 -> K=k0+8*hsel+e ; e>=8 -> K=k0+16+8*hsel+(e-8)
    v16h a;
    v8h alo = *reinterpret_cast<const v8h*>(arow + k0 + 8 * hsel);
    v8h ahi = *reinterpret_cast<const v8h*>(arow + k0 + 16 + 8 * hsel);
#pragma unroll
    for (int e = 0; e < 8; ++e) { a[e] = alo[e]; a[e + 8] = ahi[e]; }

#pragma unroll
    for (int nt = 0; nt < 4; ++nt) {
      const int nl = nt * 16 + l15;
      // ---- B fragment: lane=N, e<8 -> K=8*hsel+e ; e>=8 -> K=16+8*hsel+(e-8)
      v16h b;
      v8h blo = *reinterpret_cast<const v8h*>(&Bs[nl][8 * hsel]);
      v8h bhi = *reinterpret_cast<const v8h*>(&Bs[nl][16 + 8 * hsel]);
#pragma unroll
      for (int e = 0; e < 8; ++e) { b[e] = blo[e]; b[e + 8] = bhi[e]; }
      acc[nt] = __builtin_amdgcn_wmma_f32_16x16x32_f16(
          false, a, false, b, (short)0, acc[nt], false, false);
    }
    __syncthreads();
  }

  // ---- epilogue: C/D layout: VGPR r -> M = m0 + r + 8*hsel, N = n0 + l15
#pragma unroll
  for (int nt = 0; nt < 4; ++nt) {
    const int ncol = n0 + nt * 16 + l15;
#pragma unroll
    for (int r = 0; r < 8; ++r) {
      const int m = m0 + r + 8 * hsel;
      float v = acc[nt][r];
      if (EPI == 0) {
        ((_Float16*)Cv)[(size_t)m * N + ncol] = (_Float16)v;
      } else if (EPI == 1) {
        ((_Float16*)Cv)[(size_t)m * N + ncol] = (ncol > m) ? (_Float16)v : (_Float16)0.0f;
      } else {
        ((float*)Cv)[(size_t)m * N + ncol] = v;
      }
    }
  }
}

// ---------------- fused binary gumbel-softmax forward (in place) ----------------
__global__ void finalize(float* __restrict__ out, const float* __restrict__ ge) {
  long i = (long)blockIdx.x * blockDim.x + threadIdx.x;
  long stride = (long)gridDim.x * blockDim.x;
  for (; i < N2; i += stride) {
    float p = out[i];
    float r = 0.0f;
    if (p != 0.0f) {
      float g0 = ge[2 * i], g1 = ge[2 * i + 1];
      r = (logf(p) + g0 >= log1pf(-p) + g1) ? 1.0f : 0.0f;
    }
    out[i] = r;
  }
}

extern "C" void kernel_launch(void* const* d_in, const int* in_sizes, int n_in,
                              void* d_out, int out_size, void* d_ws, size_t ws_size,
                              hipStream_t stream) {
  const float* pw = (const float*)d_in[0];   // perm_weights [n,n] f32
  const float* ep = (const float*)d_in[1];   // edge_probs   [n,n] f32
  const float* gp = (const float*)d_in[2];   // gumbel_perm  [n,n] f32
  const float* ge = (const float*)d_in[3];   // gumbel_edge  [n,n,2] f32
  float* out = (float*)d_out;
  char* ws = (char*)d_ws;

  const size_t MB = 1ull << 20;
  float*    la = (float*)(ws);                 // 64 MB (reused: T1 @0, T3 @32MB)
  _Float16* T1 = (_Float16*)(ws);
  _Float16* T3 = (_Float16*)(ws + 32 * MB);
  _Float16* P  = (_Float16*)(ws + 64 * MB);    // 32 MB
  _Float16* PT = (_Float16*)(ws + 96 * MB);    // 32 MB
  _Float16* T2 = (_Float16*)(ws + 128 * MB);   // 32 MB
  float* R   = (float*)(ws + 160 * MB);
  float* Cv  = (float*)(ws + 160 * MB + 16384);
  int*   idx = (int*)  (ws + 160 * MB + 2 * 16384);
  float* Mp  = (float*)(ws + 160 * MB + 3 * 16384);
  float* Sp  = (float*)(ws + 160 * MB + 3 * 16384 + 32 * N * 4);
  if (ws_size < 160 * MB + 3 * 16384 + 2ull * 32 * N * 4) return;

  // la0 = pw + gumbel ; C = 0
  init_la<<<4096, 256, 0, stream>>>(pw, gp, la);
  fill_zero_f32<<<16, 256, 0, stream>>>(Cv, N);

  // 20 factored Sinkhorn iterations over L2-resident la0
  for (int t = 0; t < 20; ++t) {
    row_lse<<<N, 256, 0, stream>>>(la, Cv, R);
    col_lse_part<<<dim3(16, 32), 256, 0, stream>>>(la, R, Mp, Sp);
    col_lse_combine<<<16, 256, 0, stream>>>(Mp, Sp, Cv);
  }

  // hard permutation P (and P^T) as dense f16 matrices
  row_argmax<<<N, 256, 0, stream>>>(la, Cv, idx);
  fill_zero_u128<<<1024, 256, 0, stream>>>((uint4*)P, (64 * MB) / 16); // zero P & PT
  build_p<<<16, 256, 0, stream>>>(idx, P, PT);

  // four WMMA matmuls: probs = P^T * triu(P * E * P^T) * P
  dim3 gg(N / 64, N / 64);
  gemm_wmma<0, true ><<<gg, 128, 0, stream>>>(P,  ep, T1);   // T1 = P @ E   (E f32->f16)
  gemm_wmma<1, false><<<gg, 128, 0, stream>>>(T1, PT, T2);   // T2 = triu(T1 @ P^T)
  gemm_wmma<0, false><<<gg, 128, 0, stream>>>(PT, T2, T3);   // T3 = P^T @ T2
  gemm_wmma<2, false><<<gg, 128, 0, stream>>>(T3, P,  out);  // out = T3 @ P (f32)

  // per-edge binary gumbel-softmax forward, in place
  finalize<<<4096, 256, 0, stream>>>(out, ge);
}
